// GraphAttention_49787260895748
// MI455X (gfx1250) — matrix operand order repaired and checked by reading, not compiled, not run on hardware
//
#include <hip/hip_runtime.h>
#include <hip/hip_bf16.h>
#include <cstdint>

typedef __bf16 bf16_t;
typedef __attribute__((ext_vector_type(16))) __bf16 v16bf;
typedef __attribute__((ext_vector_type(8)))  __bf16 v8bf;
typedef __attribute__((ext_vector_type(4)))  __bf16 v4bf;
typedef __attribute__((ext_vector_type(8)))  float  v8f;
typedef __attribute__((ext_vector_type(4)))  unsigned int v4u;
typedef __attribute__((ext_vector_type(8)))  int v8i;
typedef __attribute__((ext_vector_type(4)))  int v4i;

constexpr int T_TOT = 1536;   // 512*3
constexpr int BSZ   = 8;
constexpr int EMB   = 1024;
constexpr int HEADS = 16;
constexpr int HD    = 64;     // head dim
constexpr int MROWS = T_TOT * BSZ;   // 12288
constexpr int QKVN  = 3 * EMB;       // 3072
constexpr int SEG   = 512;

__device__ __forceinline__ v8f vzero8() {
    v8f z = {0.f, 0.f, 0.f, 0.f, 0.f, 0.f, 0.f, 0.f};
    return z;
}

__device__ __forceinline__ uint32_t lds_addr_of(const void* p) {
    // generic pointers to LDS carry the LDS byte offset in the low 32 bits
    return (uint32_t)(uintptr_t)p;
}

// ---------------------------------------------------------------- TDM 2-D tile load
// Loads tile (tile1 rows x tile0 contiguous elems, bf16) from a row-major tensor
// (row stride = stride0 elements) at gsrc (tile start) into LDS at lds_byte,
// packed contiguously row-major (row stride = tile0 elements).
__device__ __forceinline__ void tdm_load_tile_2d(uint32_t lds_byte, const bf16_t* gsrc,
                                                 uint32_t tile0, uint32_t tile1,
                                                 uint32_t td0, uint32_t td1,
                                                 uint32_t stride0) {
    uint64_t ga = (uint64_t)(uintptr_t)gsrc;
    v4u g0;
    g0[0] = 1u;                                            // count=1, user descriptor
    g0[1] = lds_byte;                                      // lds_addr
    g0[2] = (uint32_t)(ga & 0xffffffffu);                  // global_addr[31:0]
    g0[3] = (uint32_t)((ga >> 32) & 0x1ffffffu) | (2u << 30); // addr[56:32] | type=2
    v8i g1;
    g1[0] = (int)(1u << 16);                               // data_size=1 (2 bytes), mask=0
    g1[1] = (int)((td0 & 0xffffu) << 16);                  // tensor_dim0[15:0]
    g1[2] = (int)((td0 >> 16) | ((td1 & 0xffffu) << 16));  // dim0[31:16] | dim1[15:0]
    g1[3] = (int)((td1 >> 16) | (tile0 << 16));            // dim1[31:16] | tile_dim0
    g1[4] = (int)(tile1 & 0xffffu);                        // tile_dim1 | tile_dim2=0
    g1[5] = (int)stride0;                                  // tensor_dim0_stride[31:0]
    g1[6] = 0;                                             // stride0 hi | stride1 lo
    g1[7] = 0;                                             // stride1 hi
    v4i g2 = {0, 0, 0, 0};
    v4i g3 = {0, 0, 0, 0};
#if defined(__clang_major__) && __clang_major__ >= 23
    v8i g4 = {0, 0, 0, 0, 0, 0, 0, 0};
    __builtin_amdgcn_tensor_load_to_lds(g0, g1, g2, g3, g4, 0);
#else
    __builtin_amdgcn_tensor_load_to_lds(g0, g1, g2, g3, 0);
#endif
}

// A-fragment (16x32 bf16, MxK): lanes 0-15 -> rows, K elems {half*8..+7, 16+half*8..+7}
__device__ __forceinline__ v16bf load_a_frag(const bf16_t* __restrict__ src, int ld, int k0) {
    int lane = threadIdx.x & 31;
    int half = lane >> 4;
    int r    = lane & 15;
    const bf16_t* base = src + (size_t)r * ld + k0 + half * 8;
    v8bf lo = *reinterpret_cast<const v8bf*>(base);       // K = half*8 .. +7
    v8bf hi = *reinterpret_cast<const v8bf*>(base + 16);  // K = 16+half*8 .. +7
    return __builtin_shufflevector(lo, hi, 0,1,2,3,4,5,6,7,8,9,10,11,12,13,14,15);
}

// B-fragment (32x16 bf16, KxN): lane r holds column N=r; K elems {half*8..+7, 16+half*8..+7}
// src is row-major [N][K-contiguous] (i.e. B^T rows)
__device__ __forceinline__ v16bf load_b_frag(const bf16_t* __restrict__ src, int ld, int k0) {
    int lane = threadIdx.x & 31;
    int half = lane >> 4;
    int r    = lane & 15;
    const bf16_t* base = src + (size_t)r * ld + k0 + half * 8;
    v8bf lo = *reinterpret_cast<const v8bf*>(base);
    v8bf hi = *reinterpret_cast<const v8bf*>(base + 16);
    return __builtin_shufflevector(lo, hi, 0,1,2,3,4,5,6,7,8,9,10,11,12,13,14,15);
}

__device__ __forceinline__ v8f wmma_bf16(v16bf a, v16bf b, v8f c) {
    return __builtin_amdgcn_wmma_f32_16x16x32_bf16(false, a, false, b, (short)0, c, false, false);
}

// ---------------------------------------------------------------- fp32 -> bf16
__global__ __launch_bounds__(256) void cvt_bf16(const float* __restrict__ src,
                                                bf16_t* __restrict__ dst, int n) {
    int i = (blockIdx.x * blockDim.x + threadIdx.x) * 4;
    if (i + 3 < n) {
        float4 v = *reinterpret_cast<const float4*>(src + i);
        v4bf o = { (bf16_t)v.x, (bf16_t)v.y, (bf16_t)v.z, (bf16_t)v.w };
        *reinterpret_cast<v4bf*>(dst + i) = o;
    }
}

// ---------------------------------------------------------------- QKV GEMM
// C[m,f] = sum_k x[m,k] * W[f,k] + b[f];  scatter into Q/K (bh,t,d) and V^T (bh,d,t)
__global__ __launch_bounds__(128) void qkv_gemm(const bf16_t* __restrict__ xw,
                                                const bf16_t* __restrict__ w1,
                                                const float*  __restrict__ bias,
                                                bf16_t* __restrict__ Q,
                                                bf16_t* __restrict__ K,
                                                bf16_t* __restrict__ Vt) {
    __shared__ __align__(16) bf16_t As[2][128 * 32];
    __shared__ __align__(16) bf16_t Bs[2][128 * 32];
    int warp = threadIdx.x >> 5;
    int lane = threadIdx.x & 31;
    int half = lane >> 4, r = lane & 15;
    int rowBlk = blockIdx.x * 128;
    int colBlk = blockIdx.y * 128;
    int wm = warp >> 1, wn = warp & 1;

    v8f acc[4][4];
#pragma unroll
    for (int mi = 0; mi < 4; mi++)
#pragma unroll
        for (int ni = 0; ni < 4; ni++) acc[mi][ni] = vzero8();

    if (warp == 0) {
        tdm_load_tile_2d(lds_addr_of(&As[0][0]), xw + (size_t)rowBlk * EMB,
                         32, 128, EMB, MROWS, EMB);
        tdm_load_tile_2d(lds_addr_of(&Bs[0][0]), w1 + (size_t)colBlk * EMB,
                         32, 128, EMB, QKVN, EMB);
    }

    for (int kt = 0; kt < EMB / 32; ++kt) {
        int cur = kt & 1;
        __syncthreads();                     // all waves done reading buffer cur^1
        if (warp == 0) {
            if (kt + 1 < EMB / 32) {
                int k0 = (kt + 1) * 32;
                tdm_load_tile_2d(lds_addr_of(&As[cur ^ 1][0]),
                                 xw + (size_t)rowBlk * EMB + k0, 32, 128, EMB, MROWS, EMB);
                tdm_load_tile_2d(lds_addr_of(&Bs[cur ^ 1][0]),
                                 w1 + (size_t)colBlk * EMB + k0, 32, 128, EMB, QKVN, EMB);
                __builtin_amdgcn_s_wait_tensorcnt((short)2);   // buffer cur complete
            } else {
                __builtin_amdgcn_s_wait_tensorcnt((short)0);
            }
        }
        __syncthreads();                     // buffer cur visible to all waves

        v16bf afr[4], bfr[4];
#pragma unroll
        for (int mi = 0; mi < 4; mi++)
            afr[mi] = load_a_frag(&As[cur][(wm * 64 + mi * 16) * 32], 32, 0);
#pragma unroll
        for (int ni = 0; ni < 4; ni++)
            bfr[ni] = load_b_frag(&Bs[cur][(wn * 64 + ni * 16) * 32], 32, 0);
#pragma unroll
        for (int ni = 0; ni < 4; ni++)
#pragma unroll
            for (int mi = 0; mi < 4; mi++)
                acc[mi][ni] = wmma_bf16(afr[mi], bfr[ni], acc[mi][ni]);
    }

    int row0 = rowBlk + wm * 64;
    int col0 = colBlk + wn * 64;
#pragma unroll
    for (int mi = 0; mi < 4; mi++) {
#pragma unroll
        for (int ni = 0; ni < 4; ni++) {
#pragma unroll
            for (int j = 0; j < 8; j++) {
                int row = row0 + mi * 16 + j + 8 * half;   // t*BSZ + b
                int col = col0 + ni * 16 + r;              // which*EMB + h*64 + dd
                float v = acc[mi][ni][j] + bias[col];
                int which = col >> 10;
                int rem   = col & 1023;
                int h  = rem >> 6;
                int dd = rem & 63;
                int t  = row >> 3;
                int b  = row & 7;
                int bh = b * HEADS + h;
                if (which == 0) {
                    Q[((size_t)bh * T_TOT + t) * HD + dd] = (bf16_t)(v * 0.125f);
                } else if (which == 1) {
                    K[((size_t)bh * T_TOT + t) * HD + dd] = (bf16_t)v;
                } else {
                    Vt[((size_t)bh * HD + dd) * T_TOT + t] = (bf16_t)v;
                }
            }
        }
    }
}

// ---------------------------------------------------------------- flash attention
// one wave = (pair p, batch-head bh, 16 q rows); keys iterated 32 at a time.
// K/V tiles TDM-prefetched into per-wave LDS double buffers.
__global__ __launch_bounds__(128) void attn_kernel(const bf16_t* __restrict__ Q,
                                                   const bf16_t* __restrict__ K,
                                                   const bf16_t* __restrict__ Vt,
                                                   bf16_t* __restrict__ attn) {
    __shared__ __align__(16) bf16_t Ks[4][2][32 * 64];
    __shared__ __align__(16) bf16_t Vs[4][2][64 * 32];
    __shared__ __align__(16) bf16_t Pbuf[4][16 * 32];
    int warp = threadIdx.x >> 5;
    int lane = threadIdx.x & 31;
    int half = lane >> 4, r = lane & 15;

    int unit = blockIdx.x * 4 + warp;      // 0 .. 3*128*32-1
    int p    = unit >> 12;                 // /4096
    int rem  = unit & 4095;
    int bh   = rem >> 5;
    int q0   = (rem & 31) * 16;
    int ki   = (p + 1) % 3;                // pairs (0,1),(1,2),(2,0)

    const bf16_t* Qseg = Q  + ((size_t)bh * T_TOT + p * SEG + q0) * HD;
    const bf16_t* Kseg = K  + ((size_t)bh * T_TOT + ki * SEG) * HD;
    const bf16_t* Vseg = Vt + (size_t)bh * HD * T_TOT + ki * SEG;

    v16bf qa0 = load_a_frag(Qseg, HD, 0);
    v16bf qa1 = load_a_frag(Qseg, HD, 32);

    v8f o[4];
#pragma unroll
    for (int ni = 0; ni < 4; ni++) o[ni] = vzero8();
    float m[8], l[8];
#pragma unroll
    for (int j = 0; j < 8; j++) { m[j] = -__builtin_inff(); l[j] = 0.f; }

    bf16_t* P = Pbuf[warp];

    // prologue prefetch: K tile (32 keys x 64d), V^T tile (64d x 32 keys)
    tdm_load_tile_2d(lds_addr_of(&Ks[warp][0][0]), Kseg, 64, 32, 64, SEG, 64);
    tdm_load_tile_2d(lds_addr_of(&Vs[warp][0][0]), Vseg, 32, 64, T_TOT, 64, T_TOT);

    for (int it = 0; it < SEG / 32; ++it) {
        int cur = it & 1;
        if (it + 1 < SEG / 32) {
            int s1 = (it + 1) * 32;
            tdm_load_tile_2d(lds_addr_of(&Ks[warp][cur ^ 1][0]),
                             Kseg + (size_t)s1 * HD, 64, 32, 64, SEG, 64);
            tdm_load_tile_2d(lds_addr_of(&Vs[warp][cur ^ 1][0]),
                             Vseg + s1, 32, 64, T_TOT, 64, T_TOT);
            __builtin_amdgcn_s_wait_tensorcnt((short)2);   // current tiles complete
        } else {
            __builtin_amdgcn_s_wait_tensorcnt((short)0);
        }

        const bf16_t* Kt = &Ks[warp][cur][0];
        v16bf kb[4];
        kb[0] = load_b_frag(Kt, 64, 0);
        kb[1] = load_b_frag(Kt, 64, 32);
        kb[2] = load_b_frag(Kt + 16 * 64, 64, 0);
        kb[3] = load_b_frag(Kt + 16 * 64, 64, 32);
        v8f sa = vzero8(), sb = vzero8();
        sa = wmma_bf16(qa0, kb[0], sa);
        sa = wmma_bf16(qa1, kb[1], sa);
        sb = wmma_bf16(qa0, kb[2], sb);
        sb = wmma_bf16(qa1, kb[3], sb);

        // V fragments: independent of softmax -> overlap ds loads with exp VALU work
        v16bf vb[4];
#pragma unroll
        for (int ni = 0; ni < 4; ni++)
            vb[ni] = load_b_frag(&Vs[warp][cur][(ni * 16) * 32], 32, 0);

#pragma unroll
        for (int j = 0; j < 8; j++) {
            float a = sa[j], b = sb[j];
            float rmax = fmaxf(a, b);
#pragma unroll
            for (int off = 1; off < 16; off <<= 1)
                rmax = fmaxf(rmax, __shfl_xor(rmax, off, 32));
            float mnew  = fmaxf(m[j], rmax);
            float scale = __expf(m[j] - mnew);
            float p0 = __expf(a - mnew);
            float p1 = __expf(b - mnew);
            float rs = p0 + p1;
#pragma unroll
            for (int off = 1; off < 16; off <<= 1)
                rs += __shfl_xor(rs, off, 32);
            l[j] = l[j] * scale + rs;
            m[j] = mnew;
            o[0][j] *= scale; o[1][j] *= scale; o[2][j] *= scale; o[3][j] *= scale;
            int prow = j + 8 * half;
            P[prow * 32 + r]      = (bf16_t)p0;
            P[prow * 32 + 16 + r] = (bf16_t)p1;
        }
        __syncthreads();
        v16bf pa = load_a_frag(P, 32, 0);    // ds_load_b128 path
#pragma unroll
        for (int ni = 0; ni < 4; ni++)
            o[ni] = wmma_bf16(pa, vb[ni], o[ni]);
        __syncthreads();
    }

    int bb = bh >> 4, h = bh & 15;
#pragma unroll
    for (int j = 0; j < 8; j++) {
        float inv = 1.0f / l[j];
        size_t rowi = (size_t)(p * SEG + q0 + j + 8 * half) * BSZ + bb;
#pragma unroll
        for (int ni = 0; ni < 4; ni++) {
            attn[rowi * EMB + h * 64 + ni * 16 + r] = (bf16_t)(o[ni][j] * inv);
        }
    }
}

// ---------------------------------------------------------------- out GEMM
__global__ __launch_bounds__(128) void out_gemm(const bf16_t* __restrict__ A,
                                                const bf16_t* __restrict__ W,
                                                const float*  __restrict__ bias,
                                                float* __restrict__ out) {
    __shared__ __align__(16) bf16_t As[2][128 * 32];
    __shared__ __align__(16) bf16_t Bs[2][128 * 32];
    int warp = threadIdx.x >> 5;
    int lane = threadIdx.x & 31;
    int half = lane >> 4, r = lane & 15;
    int rowBlk = blockIdx.x * 128;
    int colBlk = blockIdx.y * 128;
    int wm = warp >> 1, wn = warp & 1;

    v8f acc[4][4];
#pragma unroll
    for (int mi = 0; mi < 4; mi++)
#pragma unroll
        for (int ni = 0; ni < 4; ni++) acc[mi][ni] = vzero8();

    if (warp == 0) {
        tdm_load_tile_2d(lds_addr_of(&As[0][0]), A + (size_t)rowBlk * EMB,
                         32, 128, EMB, MROWS, EMB);
        tdm_load_tile_2d(lds_addr_of(&Bs[0][0]), W + (size_t)colBlk * EMB,
                         32, 128, EMB, EMB, EMB);
    }

    for (int kt = 0; kt < EMB / 32; ++kt) {
        int cur = kt & 1;
        __syncthreads();
        if (warp == 0) {
            if (kt + 1 < EMB / 32) {
                int k0 = (kt + 1) * 32;
                tdm_load_tile_2d(lds_addr_of(&As[cur ^ 1][0]),
                                 A + (size_t)rowBlk * EMB + k0, 32, 128, EMB, MROWS, EMB);
                tdm_load_tile_2d(lds_addr_of(&Bs[cur ^ 1][0]),
                                 W + (size_t)colBlk * EMB + k0, 32, 128, EMB, EMB, EMB);
                __builtin_amdgcn_s_wait_tensorcnt((short)2);
            } else {
                __builtin_amdgcn_s_wait_tensorcnt((short)0);
            }
        }
        __syncthreads();

        v16bf afr[4], bfr[4];
#pragma unroll
        for (int mi = 0; mi < 4; mi++)
            afr[mi] = load_a_frag(&As[cur][(wm * 64 + mi * 16) * 32], 32, 0);
#pragma unroll
        for (int ni = 0; ni < 4; ni++)
            bfr[ni] = load_b_frag(&Bs[cur][(wn * 64 + ni * 16) * 32], 32, 0);
#pragma unroll
        for (int ni = 0; ni < 4; ni++)
#pragma unroll
            for (int mi = 0; mi < 4; mi++)
                acc[mi][ni] = wmma_bf16(afr[mi], bfr[ni], acc[mi][ni]);
    }

    int row0 = rowBlk + wm * 64;
    int col0 = colBlk + wn * 64;
#pragma unroll
    for (int mi = 0; mi < 4; mi++) {
#pragma unroll
        for (int ni = 0; ni < 4; ni++) {
#pragma unroll
            for (int j = 0; j < 8; j++) {
                int row = row0 + mi * 16 + j + 8 * half;
                int col = col0 + ni * 16 + r;
                out[(size_t)row * EMB + col] = acc[mi][ni][j] + bias[col];
            }
        }
    }
}

extern "C" void kernel_launch(void* const* d_in, const int* in_sizes, int n_in,
                              void* d_out, int out_size, void* d_ws, size_t ws_size,
                              hipStream_t stream) {
    const float* x  = (const float*)d_in[0];   // (1536, 8, 1024)
    const float* w1 = (const float*)d_in[1];   // (3072, 1024)
    const float* b1 = (const float*)d_in[2];   // (3072,)
    const float* w2 = (const float*)d_in[3];   // (1024, 1024)
    const float* b2 = (const float*)d_in[4];   // (1024,)
    float* out = (float*)d_out;

    char* ws = (char*)d_ws;
    bf16_t* xbf  = (bf16_t*)ws;  ws += (size_t)MROWS * EMB * 2;        // 24 MiB
    bf16_t* w1bf = (bf16_t*)ws;  ws += (size_t)QKVN * EMB * 2;         //  6 MiB
    bf16_t* w2bf = (bf16_t*)ws;  ws += (size_t)EMB * EMB * 2;          //  2 MiB
    bf16_t* Qb   = (bf16_t*)ws;  ws += (size_t)BSZ * HEADS * T_TOT * HD * 2;
    bf16_t* Kb   = (bf16_t*)ws;  ws += (size_t)BSZ * HEADS * T_TOT * HD * 2;
    bf16_t* Vtb  = (bf16_t*)ws;  ws += (size_t)BSZ * HEADS * HD * T_TOT * 2;
    bf16_t* Ab   = (bf16_t*)ws;  ws += (size_t)MROWS * EMB * 2;

    {
        int n = MROWS * EMB;
        cvt_bf16<<<n / 1024, 256, 0, stream>>>(x, xbf, n);
        n = QKVN * EMB;
        cvt_bf16<<<n / 1024, 256, 0, stream>>>(w1, w1bf, n);
        n = EMB * EMB;
        cvt_bf16<<<n / 1024, 256, 0, stream>>>(w2, w2bf, n);
    }

    qkv_gemm<<<dim3(MROWS / 128, QKVN / 128), 128, 0, stream>>>(xbf, w1bf, b1, Qb, Kb, Vtb);

    attn_kernel<<<(3 * BSZ * HEADS * (SEG / 16)) / 4, 128, 0, stream>>>(Qb, Kb, Vtb, Ab);

    out_gemm<<<dim3(MROWS / 128, EMB / 128), 128, 0, stream>>>(Ab, w2bf, b2, out);
}
